// ALAttention_584115552595
// MI455X (gfx1250) — matrix operand order repaired and checked by reading, not compiled, not run on hardware
//
#include <hip/hip_runtime.h>
#include <hip/hip_bf16.h>

// ALAttention for MI455X (gfx1250, wave32, WMMA).
// Pipeline: [GEMM1 qkv, bf16 WMMA] -> [sparse attention, VALU f32] -> [GEMM2 proj, bf16 WMMA]
// GEMM2 stages its bf16 B operand with GLOBAL_LOAD_ASYNC_TO_LDS_B128 (ASYNCcnt path).

#define HWN   4096   // H*W
#define DIMC  384
#define NHEAD 6
#define HD    64
#define NBATCH 2
#define KATT  33

typedef __attribute__((ext_vector_type(16))) __bf16       v16bf;
typedef __attribute__((ext_vector_type(8)))  float        v8f;
typedef __attribute__((ext_vector_type(4)))  unsigned int v4u;

__device__ __forceinline__ unsigned int f32_to_bf16_bits(float f) {
    unsigned int u = __float_as_uint(f);
    u += 0x7FFFu + ((u >> 16) & 1u);   // round-to-nearest-even
    return u >> 16;
}
__device__ __forceinline__ unsigned int pack_bf16(float lo, float hi) {
    return f32_to_bf16_bits(lo) | (f32_to_bf16_bits(hi) << 16);
}

union Frag {
    v16bf bf;
    v4u   q[2];
};

// C[M x 4096] = A[M x K] * B[K x 4096] + bias, bf16 WMMA, f32 accumulate.
// BLAYOUT 0: B is f32 row-major [k][n]            (GEMM1: x), VALU f32->bf16 pack staging
// BLAYOUT 1: B is bf16 [k/64][n][k%64], d-contig  (GEMM2): async global->LDS staging
// EPI 0: write qkv f32 transposed [b][m/64][n][m%64], scale rows m<384 by 0.125
// EPI 1: write f32 row-major [b][m][n]            (final output)
template<int K, int BLAYOUT, int EPI>
__global__ __launch_bounds__(256)
void gemm_bf16_wmma(const float* __restrict__ A, const void* __restrict__ Bsrc,
                    const float* __restrict__ bias, float* __restrict__ dst)
{
    __shared__ __align__(16) unsigned int lds_a[128][20]; // [m][k-pair], 4-u32 row pad
    __shared__ __align__(16) unsigned int lds_b[128][20]; // [n][k-pair]

    const int tid   = threadIdx.x;
    const int b     = blockIdx.z;
    const int m_blk = blockIdx.y * 128;
    const int n_blk = blockIdx.x * 128;

    const int wave = tid >> 5;
    const int lane = tid & 31;
    const int half = lane >> 4;    // 0: lanes 0-15, 1: lanes 16-31
    const int lc   = lane & 15;
    const int wm   = wave & 3;     // 4 m-groups of 32 rows
    const int wn   = wave >> 2;    // 2 n-groups of 64 cols

    const float*          Bf = (const float*)Bsrc          + (size_t)b * K * HWN;
    const unsigned short* Bh = (const unsigned short*)Bsrc + (size_t)b * K * HWN;

    v8f acc[2][4] = {};

    const int a_pc = tid & 15,  a_mr = tid >> 4;   // A stage: 16 pairs x 16 rows, x8 iters
    const int b0_n = tid & 127, b0_kh = tid >> 7;  // layout0 stage
    const int b1_c4 = tid & 3,  b1_nr = tid >> 2;  // layout1 stage: 16B chunk, n-row

    for (int kt = 0; kt < K / 32; ++kt) {
        const int k0 = kt * 32;

        // gfx1250 prefetch of next K-tile (global_prefetch_b8)
        if (kt + 1 < K / 32) {
            __builtin_prefetch(&A[(size_t)(m_blk + a_mr) * K + k0 + 32 + 2 * a_pc], 0, 3);
            if (BLAYOUT == 0)
                __builtin_prefetch(&Bf[(size_t)(k0 + 32 + b0_kh) * HWN + n_blk + b0_n], 0, 3);
        }

        // stage A: 128x32 f32 weights -> bf16 pairs, coalesced 8B/thread
        #pragma unroll
        for (int i = 0; i < 8; ++i) {
            int m = a_mr + i * 16;
            size_t off = (size_t)(m_blk + m) * K + k0 + 2 * a_pc;
            lds_a[m][a_pc] = pack_bf16(A[off], A[off + 1]);
        }

        // stage B -> LDS [n][k-pair] (fragment-friendly: per-lane fixed n, K contiguous)
        if (BLAYOUT == 0) {
            #pragma unroll
            for (int i = 0; i < 8; ++i) {
                int k2 = b0_kh + 2 * i;
                int kg = k0 + 2 * k2;
                float x0 = Bf[(size_t)kg * HWN + n_blk + b0_n];
                float x1 = Bf[(size_t)(kg + 1) * HWN + n_blk + b0_n];
                lds_b[b0_n][k2] = pack_bf16(x0, x1);
            }
        } else {
            // bf16 source already tile-shaped: each n-row's 16 k-pairs are one
            // contiguous 64B global segment -> pure copy via the CDNA5 async path.
            const int hh  = k0 >> 6;   // head block (K-tile never straddles a head)
            const int dlo = k0 & 63;   // d offset of this k-tile within the head
            #pragma unroll
            for (int i = 0; i < 2; ++i) {
                int n = b1_nr + 64 * i;
                const unsigned short* gp =
                    Bh + (((size_t)hh * HWN + n_blk + n) * HD + dlo + b1_c4 * 8);
                unsigned int lds_addr =
                    (unsigned int)(unsigned long long)(const void*)&lds_b[n][b1_c4 * 4];
                unsigned long long ga = (unsigned long long)gp;
                asm volatile("global_load_async_to_lds_b128 %0, %1, off"
                             :: "v"(lds_addr), "v"(ga) : "memory");
            }
#if __has_builtin(__builtin_amdgcn_s_wait_asynccnt)
            __builtin_amdgcn_s_wait_asynccnt(0);
#else
            asm volatile("s_wait_asynccnt 0" ::: "memory");
#endif
        }
        __syncthreads();

        // load fragments per documented 16-bit WMMA VGPR layouts
        Frag fa[2], fb[4];
        #pragma unroll
        for (int tm = 0; tm < 2; ++tm) {
            const unsigned int* p = &lds_a[wm * 32 + tm * 16 + lc][half * 4];
            fa[tm].q[0] = *(const v4u*)p;        // K 0..7 / 8..15
            fa[tm].q[1] = *(const v4u*)(p + 8);  // K 16..23 / 24..31
        }
        #pragma unroll
        for (int tn = 0; tn < 4; ++tn) {
            const unsigned int* p = &lds_b[wn * 64 + tn * 16 + lc][half * 4];
            fb[tn].q[0] = *(const v4u*)p;
            fb[tn].q[1] = *(const v4u*)(p + 8);
        }

        #pragma unroll
        for (int tm = 0; tm < 2; ++tm)
            #pragma unroll
            for (int tn = 0; tn < 4; ++tn)
                acc[tm][tn] = __builtin_amdgcn_wmma_f32_16x16x32_bf16(
                    false, fa[tm].bf, false, fb[tn].bf, (short)0, acc[tm][tn], false, false);
        __syncthreads();
    }

    // epilogue
    #pragma unroll
    for (int tm = 0; tm < 2; ++tm) {
        #pragma unroll
        for (int tn = 0; tn < 4; ++tn) {
            const int m_t = m_blk + wm * 32 + tm * 16;
            const int n_t = n_blk + wn * 64 + tn * 16;
            float r[8];
            #pragma unroll
            for (int i = 0; i < 8; ++i)
                r[i] = acc[tm][tn][i] + bias[m_t + half * 8 + i];
            if (EPI == 0) {
                const float scale = (m_t < DIMC) ? 0.125f : 1.0f; // q rows pre-scaled
                #pragma unroll
                for (int i = 0; i < 8; ++i) r[i] *= scale;
                const int sh = m_t >> 6;                 // (s*6+h) in [0,18)
                const int d0 = (m_t & 63) + half * 8;
                float* dp = dst + (((size_t)(b * 18 + sh) * HWN) + n_t + lc) * HD + d0;
                *(float4*)dp       = make_float4(r[0], r[1], r[2], r[3]);
                *(float4*)(dp + 4) = make_float4(r[4], r[5], r[6], r[7]);
            } else {
                float* dp = dst + ((size_t)b * DIMC + m_t + half * 8) * HWN + n_t + lc;
                #pragma unroll
                for (int i = 0; i < 8; ++i) dp[(size_t)i * HWN] = r[i];
            }
        }
    }
}

// One wave32 per query row: 33 gathered keys -> scores -> softmax -> weighted V sum.
__global__ __launch_bounds__(256)
void attn_kernel(const float* __restrict__ qkv, const int* __restrict__ attn_idx,
                 unsigned short* __restrict__ attn16)
{
    __shared__ float sc[8][40];
    const int tid  = threadIdx.x;
    const int wv   = tid >> 5;
    const int lane = tid & 31;
    const int qid  = blockIdx.x * 8 + wv;
    const int n  = qid & (HWN - 1);
    const int bh = qid >> 12;           // b*6+h
    const int h  = bh % NHEAD;
    const int b  = bh / NHEAD;

    const float* qp = qkv + (((size_t)(b * 18 + h) * HWN) + n) * HD;
    const float* kb = qkv + ((size_t)(b * 18 + 6  + h) * HWN) * HD;
    const float* vb = qkv + ((size_t)(b * 18 + 12 + h) * HWN) * HD;
    const int* idx = attn_idx + (size_t)n * KATT;

    const float qx = qp[lane], qy = qp[lane + 32];   // q already scaled by 0.125

    for (int k = 0; k < KATT; ++k) {
        const float* kr = kb + (size_t)idx[k] * HD;   // coalesced 256B row gather
        float p = qx * kr[lane] + qy * kr[lane + 32];
        #pragma unroll
        for (int off = 16; off; off >>= 1) p += __shfl_xor(p, off);
        if (lane == 0) sc[wv][k] = p;
    }
    __syncthreads();

    float s   = sc[wv][lane];
    float s32 = sc[wv][32];
    float m = s;
    #pragma unroll
    for (int off = 16; off; off >>= 1) m = fmaxf(m, __shfl_xor(m, off));
    m = fmaxf(m, s32);
    float e   = __expf(s - m);
    float e32 = __expf(s32 - m);
    float t = e;
    #pragma unroll
    for (int off = 16; off; off >>= 1) t += __shfl_xor(t, off);
    t += e32;
    const float inv = 1.0f / t;
    sc[wv][lane] = e * inv;
    if (lane == 0) sc[wv][32] = e32 * inv;
    __syncthreads();

    float ax = 0.f, ay = 0.f;
    for (int k = 0; k < KATT; ++k) {
        const float p = sc[wv][k];                    // LDS broadcast
        const float* vr = vb + (size_t)idx[k] * HD;
        ax += p * vr[lane];
        ay += p * vr[lane + 32];
    }
    // bf16 output in [b][h][n][d] -> exactly K-contiguous for GEMM2 B fragments
    unsigned short* op = attn16 + (((size_t)(b * NHEAD + h) * HWN) + n) * HD;
    op[lane]      = (unsigned short)f32_to_bf16_bits(ax);
    op[lane + 32] = (unsigned short)f32_to_bf16_bits(ay);
}

extern "C" void kernel_launch(void* const* d_in, const int* in_sizes, int n_in,
                              void* d_out, int out_size, void* d_ws, size_t ws_size,
                              hipStream_t stream) {
    (void)in_sizes; (void)n_in; (void)out_size; (void)ws_size;
    const float* x      = (const float*)d_in[0];
    const float* w_qkv  = (const float*)d_in[1];
    const float* b_qkv  = (const float*)d_in[2];
    const float* w_proj = (const float*)d_in[3];
    const float* b_proj = (const float*)d_in[4];
    const int*   aidx   = (const int*)d_in[5];
    float* out = (float*)d_out;

    // workspace: qkv f32 [B][18][4096][64] (37.75MB) + attn out bf16 [B][6][4096][64] (6.3MB)
    const size_t QKV_BYTES = (size_t)NBATCH * 18 * HWN * HD * sizeof(float);
    float* qkv = (float*)d_ws;
    unsigned short* attn16 = (unsigned short*)((char*)d_ws + QKV_BYTES);

    // GEMM1: M=1152, K=384, N=4096, per batch
    gemm_bf16_wmma<DIMC, 0, 0><<<dim3(HWN / 128, (3 * DIMC) / 128, NBATCH), 256, 0, stream>>>(
        w_qkv, x, b_qkv, qkv);

    // Attention: one wave per query, 8 queries per block
    attn_kernel<<<dim3((NBATCH * NHEAD * HWN) / 8), 256, 0, stream>>>(qkv, aidx, attn16);

    // GEMM2: M=384, K=384, N=4096, per batch
    gemm_bf16_wmma<DIMC, 1, 1><<<dim3(HWN / 128, DIMC / 128, NBATCH), 256, 0, stream>>>(
        w_proj, attn16, b_proj, out);
}
